// UltimatexLSTMBlock_3006477107701
// MI455X (gfx1250) — compile-verified, matrix operand order checked by reading, not run on hardware
//
#include <hip/hip_runtime.h>
#include <cstdint>
#include <cstddef>

// ---------------------------------------------------------------------------
// Problem constants (from reference)
// ---------------------------------------------------------------------------
#define BB      4
#define SS      2048
#define TT      (BB * SS)          // 8192 tokens
#define INP     512
#define HEADS   8
#define HD      64
#define HID     512                // HEADS*HD
#define PROJ    1024               // 2*INP
#define NCAT    1536               // PROJ + HID  (W_up || W_r)
#define NF      2064               // 2*HEADS + 4*HID (live columns)
#define NFP     2112               // NF padded to a multiple of 64
#define CAPV    15.0f

typedef __attribute__((ext_vector_type(16))) __bf16    v16bf;
typedef __attribute__((ext_vector_type(8)))  float     v8f;
typedef __attribute__((ext_vector_type(4)))  unsigned  v4u;

// Swizzle of the K index inside a 32-wide K tile so that each lane's WMMA
// fragment ({0..7,16..23} for lanes 0-15, {8..15,24..31} for lanes 16-31)
// becomes 16 contiguous bf16.  perm swaps bits 3 and 4.
__device__ __forceinline__ int kperm(int k) {
    return (k & 7) | ((k & 8) << 1) | ((k & 16) >> 1);
}

// Load a 32-byte fragment from LDS as two 16-byte chunks (16B alignment
// guaranteed by the row stride of 40 elements = 80 bytes).
__device__ __forceinline__ v16bf ld_frag(const __bf16* p) {
    union { v4u u[2]; v16bf v; } t;
    t.u[0] = *(const v4u*)(p);
    t.u[1] = *(const v4u*)(p + 8);
    return t.v;
}

// ---------------------------------------------------------------------------
// Weight prep: fp32 row-major [K][cols] -> bf16 fragment layout
//   Wp[k/32][col][perm(k%32)]   (zero-padded to cd columns)
// so GEMM B-staging is straight 16B global->LDS copies, no scatter.
// ---------------------------------------------------------------------------
__global__ void pack_wfrag_kernel(const float* __restrict__ src,
                                  __bf16* __restrict__ dst,
                                  int KD, int cs, int cd)
{
    int idx = blockIdx.x * blockDim.x + threadIdx.x;
    int total = KD * cd;
    for (; idx < total; idx += gridDim.x * blockDim.x) {
        int k = idx / cd;
        int n = idx - k * cd;
        float v = (n < cs) ? src[(size_t)k * cs + n] : 0.0f;
        dst[((size_t)(k >> 5) * cd + n) * 32 + kperm(k & 31)] = (__bf16)v;
    }
}

// [W_up | W_r] concat, straight into fragment layout
__global__ void pack_wcat_frag_kernel(const float* __restrict__ Wup,
                                      const float* __restrict__ Wr,
                                      __bf16* __restrict__ dst)
{
    int idx = blockIdx.x * blockDim.x + threadIdx.x;
    int total = INP * NCAT;
    for (; idx < total; idx += gridDim.x * blockDim.x) {
        int k = idx / NCAT;
        int n = idx - k * NCAT;
        float v = (n < PROJ) ? Wup[(size_t)k * PROJ + n]
                             : Wr[(size_t)k * HID + (n - PROJ)];
        dst[((size_t)(k >> 5) * NCAT + n) * 32 + kperm(k & 31)] = (__bf16)v;
    }
}

__global__ void pad_bias_kernel(const float* __restrict__ src,
                                float* __restrict__ dst, int ns, int nd)
{
    int idx = blockIdx.x * blockDim.x + threadIdx.x;
    for (; idx < nd; idx += gridDim.x * blockDim.x)
        dst[idx] = (idx < ns) ? src[idx] : 0.0f;
}

// ---------------------------------------------------------------------------
// Input LayerNorm -> bf16 activations (one block per token, 1 float4/thread)
// ---------------------------------------------------------------------------
__global__ __launch_bounds__(128)
void ln_in_kernel(const float* __restrict__ x, const float* __restrict__ g,
                  const float* __restrict__ b, __bf16* __restrict__ out)
{
    __shared__ float s1[128], s2[128];
    const int t   = blockIdx.x;
    const int tid = threadIdx.x;
    const float* xr = x + (size_t)t * INP;

    const float4 v4 = *(const float4*)(xr + tid * 4);
    float a0 = v4.x + v4.y + v4.z + v4.w;
    float a1 = v4.x * v4.x + v4.y * v4.y + v4.z * v4.z + v4.w * v4.w;
    s1[tid] = a0; s2[tid] = a1;
    __syncthreads();
    for (int off = 64; off > 0; off >>= 1) {
        if (tid < off) { s1[tid] += s1[tid + off]; s2[tid] += s2[tid + off]; }
        __syncthreads();
    }
    const float mu   = s1[0] * (1.0f / INP);
    const float var  = s2[0] * (1.0f / INP) - mu * mu;
    const float rstd = rsqrtf(var + 1e-6f);

    const float4 g4 = *(const float4*)(g + tid * 4);
    const float4 b4 = *(const float4*)(b + tid * 4);
    __bf16* op = out + (size_t)t * INP + tid * 4;
    op[0] = (__bf16)((v4.x - mu) * rstd * g4.x + b4.x);
    op[1] = (__bf16)((v4.y - mu) * rstd * g4.y + b4.y);
    op[2] = (__bf16)((v4.z - mu) * rstd * g4.z + b4.z);
    op[3] = (__bf16)((v4.w - mu) * rstd * g4.w + b4.w);
}

// ---------------------------------------------------------------------------
// bf16 x bf16 -> fp32 tiled GEMM via v_wmma_f32_16x16x32_bf16
//   Out[M, NCOLS] = A[M, KDIM] @ W (+bias) (+resid)
//   A: row-major bf16 [M][KDIM].  W: fragment layout [KDIM/32][NCOLS][32].
//   Compile-time NCOLS (mult of 128) / KDIM (mult of 64): no guards.
//   128 threads = 4 waves; block tile 64x128; each wave owns a 32x64 subtile
//   (2x4 WMMA accumulators).  All staging is 16B vector copies.  Double-
//   buffered LDS pipeline: global->regs for tile k+1 overlaps WMMA on tile k.
// ---------------------------------------------------------------------------
#define LDST 40   // LDS row stride in elements (80B: 16B-aligned, conflict-light)
#define ASZ  (64 * LDST)
#define BSZ  (128 * LDST)

template <int NCOLS, int KDIM, bool HAS_BIAS, bool HAS_RES>
__global__ __launch_bounds__(128)
void gemm_bf16_wmma(const __bf16* __restrict__ A, const __bf16* __restrict__ W,
                    float* __restrict__ Out, const float* __restrict__ bias,
                    const float* __restrict__ resid)
{
    __shared__ __attribute__((aligned(32))) __bf16 As2[2 * ASZ]; // [buf][row][fragk]
    __shared__ __attribute__((aligned(32))) __bf16 Bs2[2 * BSZ]; // [buf][col][fragk]

    const int tid  = threadIdx.x;
    const int lane = tid & 31;
    const int wv   = tid >> 5;            // wave 0..3
    const int wr   = wv >> 1;             // wave row (0..1) -> 32 rows
    const int wc   = wv & 1;              // wave col (0..1) -> 64 cols
    const int m0   = blockIdx.y * 64;
    const int n0   = blockIdx.x * 128;
    const int lr   = lane & 15;
    const int kbo  = (lane < 16) ? 0 : 16;  // fragment offset after swizzle

    // ---- A staging: 64x32 bf16/tile, two 16B chunks per thread
    const int ar  = tid >> 2;                          // 0..31
    const int ag  = tid & 3;                           // K group of 8
    const int apg = ((ag & 1) << 1) | ((ag & 2) >> 1); // permuted group
    const __bf16* aP0 = A + (size_t)(m0 + ar) * KDIM + ag * 8;
    const __bf16* aP1 = A + (size_t)(m0 + 32 + ar) * KDIM + ag * 8;
    const int aD0 = ar * LDST + apg * 8;
    const int aD1 = (32 + ar) * LDST + apg * 8;

    // ---- B staging: 128 cols x 32 fragk bf16/tile = 512 16B chunks, 4/thread
    const __bf16* bP[4];
    int bD[4];
    #pragma unroll
    for (int j = 0; j < 4; ++j) {
        const int e    = tid + j * 128;   // 0..511
        const int col  = e >> 2;          // 0..127
        const int part = e & 3;           // 8-elem group within fragment
        bP[j] = W + ((size_t)(n0 + col)) * 32 + part * 8;
        bD[j] = col * LDST + part * 8;
    }

    v4u ra0, ra1, rb[4];
    auto load_tile = [&]() {
        ra0 = *(const v4u*)aP0;  aP0 += 32;
        ra1 = *(const v4u*)aP1;  aP1 += 32;
        #pragma unroll
        for (int j = 0; j < 4; ++j) { rb[j] = *(const v4u*)bP[j]; bP[j] += (size_t)NCOLS * 32; }
    };
    auto store_tile = [&](int buf) {
        __bf16* ab = As2 + buf * ASZ;
        __bf16* bb = Bs2 + buf * BSZ;
        *(v4u*)&ab[aD0] = ra0;
        *(v4u*)&ab[aD1] = ra1;
        #pragma unroll
        for (int j = 0; j < 4; ++j)
            *(v4u*)&bb[bD[j]] = rb[j];
    };

    v8f acc[2][4] = {};
    auto compute = [&](int buf) {
        const __bf16* ab = As2 + buf * ASZ;
        const __bf16* bb = Bs2 + buf * BSZ;
        v16bf af[2], bfv[4];
        #pragma unroll
        for (int mt = 0; mt < 2; ++mt)
            af[mt] = ld_frag(&ab[(wr * 32 + mt * 16 + lr) * LDST + kbo]);
        #pragma unroll
        for (int nt = 0; nt < 4; ++nt)
            bfv[nt] = ld_frag(&bb[(wc * 64 + nt * 16 + lr) * LDST + kbo]);
        #pragma unroll
        for (int mt = 0; mt < 2; ++mt)
            #pragma unroll
            for (int nt = 0; nt < 4; ++nt)
                acc[mt][nt] = __builtin_amdgcn_wmma_f32_16x16x32_bf16(
                    false, af[mt], false, bfv[nt],
                    (short)0, acc[mt][nt], false, false);
    };

    // ---- pipelined main loop: 1 barrier per K step
    load_tile();
    store_tile(0);
    __syncthreads();
    int cur = 0;
    for (int k0 = 32; k0 < KDIM; k0 += 32) {
        __builtin_prefetch(aP0 + 32, 0, 1);   // next-next A tile (global_prefetch_b8)
        load_tile();                          // global -> regs (tile k0)
        compute(cur);                         // WMMA on tile k0-32
        store_tile(cur ^ 1);                  // regs -> LDS (tile k0)
        __syncthreads();
        cur ^= 1;
    }
    compute(cur);

    // ---- epilogue: C/D layout => VGPR i: lanes 0-15 M=i, lanes 16-31 M=8+i
    #pragma unroll
    for (int mt = 0; mt < 2; ++mt) {
        #pragma unroll
        for (int nt = 0; nt < 4; ++nt) {
            const int nn = n0 + wc * 64 + nt * 16 + lr;
            const float bv = HAS_BIAS ? bias[nn] : 0.0f;
            #pragma unroll
            for (int i = 0; i < 8; ++i) {
                const int mm = m0 + wr * 32 + mt * 16 + ((lane < 16) ? i : 8 + i);
                float v = acc[mt][nt][i] + bv;
                if (HAS_RES) v += resid[(size_t)mm * NCOLS + nn];
                Out[(size_t)mm * NCOLS + nn] = v;
            }
        }
    }
}

// ---------------------------------------------------------------------------
// Causal 4-tap conv along the feature axis + SiLU -> bf16
// ---------------------------------------------------------------------------
__global__ void conv_silu_kernel(const float* __restrict__ upr,
                                 const float* __restrict__ cw,
                                 __bf16* __restrict__ xconv)
{
    int idx = blockIdx.x * blockDim.x + threadIdx.x;
    const int total = TT * PROJ;
    for (; idx < total; idx += gridDim.x * blockDim.x) {
        const int t = idx >> 10;            // /PROJ
        const int p = idx & (PROJ - 1);
        const float* row = upr + (size_t)t * NCAT;
        float s = 0.f;
        #pragma unroll
        for (int j = 0; j < 4; ++j) {
            int pp = p - 3 + j;
            if (pp >= 0) s += cw[j] * row[pp];
        }
        float sv = s / (1.f + __expf(-s));  // silu
        xconv[idx] = (__bf16)sv;
    }
}

// ---------------------------------------------------------------------------
// mLSTM sequential scan. One block per (b,h); 256 threads.
// Thread tid owns row r = tid>>2, cols [16*(tid&3), +15] of 64x64 state C.
// ---------------------------------------------------------------------------
__global__ __launch_bounds__(256)
void mlstm_scan_kernel(const float* __restrict__ fused,
                       float* __restrict__ g_out,
                       float* __restrict__ state_out)
{
    const int bh  = blockIdx.x;            // 0..31
    const int b   = bh >> 3;
    const int h   = bh & 7;
    const int tid = threadIdx.x;
    const int r   = tid >> 2;              // 0..63
    const int qd  = tid & 3;
    const int c0  = qd * 16;

    __shared__ float q_s[64], k_s[64], v_s[64], n_s[64], dp[64];
    __shared__ float red[64][4];
    __shared__ float fe_s, ie_s, den_s;

    float C[16];
    #pragma unroll
    for (int j = 0; j < 16; ++j) C[j] = 0.f;
    if (tid < 64) n_s[tid] = 1.0f;
    float m = 0.0f;                        // only thread 0's copy matters

    for (int s = 0; s < SS; ++s) {
        const size_t tok = (size_t)b * SS + s;
        const float* fr  = fused + tok * NFP;

        __syncthreads();                   // A: previous consumers done
        float o_val = 0.f;
        if (tid < 64) {
            q_s[tid] = fr[16 + h * HD + tid];
            k_s[tid] = fr[16 + HID + h * HD + tid] * 0.125f;  // 1/sqrt(64)
            v_s[tid] = fr[16 + 2 * HID + h * HD + tid];
            o_val    = fr[16 + 3 * HID + h * HD + tid];
        }
        if (tid == 0) {
            float iv = fr[h], fv = fr[HEADS + h];
            iv = CAPV * tanhf(iv / CAPV);
            fv = CAPV * tanhf(fv / CAPV);
            float m_t = fmaxf(fv + m, iv);
            fe_s = __expf(fv - m_t + m);
            ie_s = __expf(iv - m_t);
            m = m_t;
        }
        __syncthreads();                   // B: gates + qkv visible

        const float fe = fe_s, ie = ie_s;
        const float vr = v_s[r];
        float part = 0.f;
        #pragma unroll
        for (int j = 0; j < 16; ++j) {
            const int c = c0 + j;
            C[j] = fe * C[j] + ie * vr * k_s[c];
            part += C[j] * q_s[c];
        }
        red[r][qd] = part;
        if (tid < 64) {
            float nn = fe * n_s[tid] + ie * k_s[tid];
            n_s[tid] = nn;
            dp[tid]  = nn * q_s[tid];
        }
        __syncthreads();                   // C: partials + dp visible
        if (tid == 0) {
            float dsum = 0.f;
            #pragma unroll 8
            for (int j = 0; j < 64; ++j) dsum += dp[j];
            den_s = fmaxf(dsum, 1.0f);
        }
        __syncthreads();                   // D: denominator visible
        if (tid < 64) {
            float hn = red[tid][0] + red[tid][1] + red[tid][2] + red[tid][3];
            float og = 1.f / (1.f + __expf(-o_val));
            g_out[tok * HID + h * HD + tid] = (hn / den_s) * og;
        }
    }

    // final states: C_f (B,H,64,64), n_f (B,H,64), m_f (B,H)
    __syncthreads();
    float* Cf = state_out;
    float* nf = state_out + (size_t)BB * HEADS * HD * HD;
    float* mf = nf + (size_t)BB * HEADS * HD;
    #pragma unroll
    for (int j = 0; j < 16; ++j)
        Cf[((size_t)bh * HD + r) * HD + c0 + j] = C[j];
    if (tid < 64) nf[(size_t)bh * HD + tid] = n_s[tid];
    if (tid == 0) mf[bh] = m;
}

// ---------------------------------------------------------------------------
// LN over HID of g, times silu(r)  -> bf16 (one block per token)
// ---------------------------------------------------------------------------
__global__ __launch_bounds__(128)
void ln_h_kernel(const float* __restrict__ gin, const float* __restrict__ hg,
                 const float* __restrict__ hb, const float* __restrict__ upr,
                 __bf16* __restrict__ y)
{
    __shared__ float s1[128], s2[128];
    const int t   = blockIdx.x;
    const int tid = threadIdx.x;
    const float* xr = gin + (size_t)t * HID;

    const float4 v4 = *(const float4*)(xr + tid * 4);
    float a0 = v4.x + v4.y + v4.z + v4.w;
    float a1 = v4.x * v4.x + v4.y * v4.y + v4.z * v4.z + v4.w * v4.w;
    s1[tid] = a0; s2[tid] = a1;
    __syncthreads();
    for (int off = 64; off > 0; off >>= 1) {
        if (tid < off) { s1[tid] += s1[tid + off]; s2[tid] += s2[tid + off]; }
        __syncthreads();
    }
    const float mu   = s1[0] * (1.0f / HID);
    const float var  = s2[0] * (1.0f / HID) - mu * mu;
    const float rstd = rsqrtf(var + 1e-6f);

    const float4 g4 = *(const float4*)(hg + tid * 4);
    const float4 b4 = *(const float4*)(hb + tid * 4);
    const float4 r4 = *(const float4*)(upr + (size_t)t * NCAT + PROJ + tid * 4);
    float ln0 = (v4.x - mu) * rstd * g4.x + b4.x;
    float ln1 = (v4.y - mu) * rstd * g4.y + b4.y;
    float ln2 = (v4.z - mu) * rstd * g4.z + b4.z;
    float ln3 = (v4.w - mu) * rstd * g4.w + b4.w;
    __bf16* op = y + (size_t)t * HID + tid * 4;
    op[0] = (__bf16)(ln0 * (r4.x / (1.f + __expf(-r4.x))));
    op[1] = (__bf16)(ln1 * (r4.y / (1.f + __expf(-r4.y))));
    op[2] = (__bf16)(ln2 * (r4.z / (1.f + __expf(-r4.z))));
    op[3] = (__bf16)(ln3 * (r4.w / (1.f + __expf(-r4.w))));
}

// ---------------------------------------------------------------------------
// Launch
// ---------------------------------------------------------------------------
extern "C" void kernel_launch(void* const* d_in, const int* in_sizes, int n_in,
                              void* d_out, int out_size, void* d_ws, size_t ws_size,
                              hipStream_t stream)
{
    (void)in_sizes; (void)n_in; (void)out_size; (void)ws_size;

    const float* x       = (const float*)d_in[0];
    const float* inp_g   = (const float*)d_in[1];
    const float* inp_b   = (const float*)d_in[2];
    const float* hid_g   = (const float*)d_in[3];
    const float* hid_b   = (const float*)d_in[4];
    const float* W_up    = (const float*)d_in[5];
    const float* W_r     = (const float*)d_in[6];
    const float* W_fused = (const float*)d_in[7];
    const float* b_fused = (const float*)d_in[8];
    const float* W_down  = (const float*)d_in[9];
    const float* conv_w  = (const float*)d_in[10];

    float* out = (float*)d_out;

    // workspace carve-up (byte offsets, 256B aligned)
    char* ws = (char*)d_ws;
    size_t off = 0;
    auto carve = [&](size_t bytes) -> char* {
        char* p = ws + off;
        off += (bytes + 255) & ~(size_t)255;
        return p;
    };
    __bf16* xn    = (__bf16*)carve((size_t)TT * INP * 2);      // LN(x) bf16
    __bf16* Wcat  = (__bf16*)carve((size_t)INP * NCAT * 2);    // frag [W_up|W_r]
    __bf16* Wf    = (__bf16*)carve((size_t)PROJ * NFP * 2);    // frag W_fused (pad)
    __bf16* Wd    = (__bf16*)carve((size_t)HID * INP * 2);     // frag W_down
    float*  bfp   = (float*)carve((size_t)NFP * 4);            // padded bias
    float*  upr   = (float*)carve((size_t)TT * NCAT * 4);      // [x_up | r]
    __bf16* xconv = (__bf16*)carve((size_t)TT * PROJ * 2);     // silu(conv)
    float*  fused = (float*)carve((size_t)TT * NFP * 4);       // gate pre-acts
    float*  gbuf  = (float*)carve((size_t)TT * HID * 4);       // h * sigmoid(o)
    __bf16* ybuf  = (__bf16*)carve((size_t)TT * HID * 2);      // LN(h)*silu(r)

    // 1) weights -> bf16 fragment layout (B-staging becomes pure 16B copies)
    pack_wcat_frag_kernel<<<(INP * NCAT + 255) / 256, 256, 0, stream>>>(W_up, W_r, Wcat);
    pack_wfrag_kernel<<<(PROJ * NFP + 255) / 256, 256, 0, stream>>>(
        W_fused, Wf, PROJ, NF, NFP);
    pack_wfrag_kernel<<<(HID * INP + 255) / 256, 256, 0, stream>>>(
        W_down, Wd, HID, INP, INP);
    pad_bias_kernel<<<(NFP + 255) / 256, 256, 0, stream>>>(b_fused, bfp, NF, NFP);

    // 2) LN(x) -> bf16
    ln_in_kernel<<<TT, 128, 0, stream>>>(x, inp_g, inp_b, xn);

    // 3) [x_up | r] = xn @ [W_up | W_r]   (T x 1536, K=512)
    gemm_bf16_wmma<NCAT, INP, false, false>
        <<<dim3(NCAT / 128, TT / 64), 128, 0, stream>>>(
            xn, Wcat, upr, nullptr, nullptr);

    // 4) conv + silu -> bf16
    conv_silu_kernel<<<(TT * PROJ + 255) / 256, 256, 0, stream>>>(upr, conv_w, xconv);

    // 5) fused = x_conv @ W_fused + b_fused   (T x 2112(pad), K=1024)
    gemm_bf16_wmma<NFP, PROJ, true, false>
        <<<dim3(NFP / 128, TT / 64), 128, 0, stream>>>(
            xconv, Wf, fused, bfp, nullptr);

    // 6) mLSTM scan -> g, final states straight into d_out tail
    mlstm_scan_kernel<<<BB * HEADS, 256, 0, stream>>>(
        fused, gbuf, out + (size_t)TT * INP);

    // 7) y = LN(g) * silu(r) -> bf16
    ln_h_kernel<<<TT, 128, 0, stream>>>(gbuf, hid_g, hid_b, upr, ybuf);

    // 8) out = y @ W_down + x   (T x 512, K=512)
    gemm_bf16_wmma<INP, HID, false, true>
        <<<dim3(INP / 128, TT / 64), 128, 0, stream>>>(
            ybuf, Wd, out, nullptr, x);
}
